// BRFModel_46909632807511
// MI455X (gfx1250) — compile-verified, matrix operand order checked by reading, not run on hardware
//
#include <hip/hip_runtime.h>

typedef _Float16 v16h __attribute__((ext_vector_type(16)));
typedef _Float16 v8h  __attribute__((ext_vector_type(8)));
typedef float    v8f  __attribute__((ext_vector_type(8)));

namespace {
constexpr int H = 2400, W = 2400, SX = 30, SY = 30;
constexpr int NBX = H / SX, NBY = W / SY;   // 80 x 80 blocks
constexpr int NBLK = NBX * NBY;             // 6400
constexpr int NPIX = SX * SY;               // 900 pixels per block
constexpr float Gc  = 0.5f;
constexpr float PIf = 3.14159265358979323846f;
constexpr float DEG = PIf / 180.0f;
// workspace layout: ws[0] = global max (float bits via int atomicMax)
//                   ws[1 .. NBLK]          = A term per block
//                   ws[1+NBLK .. 2*NBLK]   = B term per block (needs /max)
}

__global__ __launch_bounds__(256) void ws_init_kernel(float* ws) {
  if (threadIdx.x == 0) ws[0] = 0.0f;   // 0.0f bits; positive-float atomicMax on int bits
}

// Single pass over CHM: halo stencil in LDS, 7-channel block reduce via WMMA,
// per-block CHM max -> atomicMax, partial BRF (A, B) to workspace.
__global__ __launch_bounds__(256) void brf_kernel(
    const float* __restrict__ chm,  const float* __restrict__ TH,
    const float* __restrict__ FAVD, const float* __restrict__ sza,
    const float* __restrict__ saa,  const float* __restrict__ rl,
    const float* __restrict__ tl,   const float* __restrict__ rs,
    const float* __restrict__ belta, float* __restrict__ ws) {
  __shared__ float tile[32][33];                 // 30x30 block + 1-pixel halo
  __shared__ float part[7][260];                 // per-thread f32 partials
  __shared__ float mred[256];                    // block-max reduction
  __shared__ alignas(16) _Float16 part2[16][32]; // compacted matrix for WMMA

  const int tid = threadIdx.x;
  const int bx = blockIdx.x, by = blockIdx.y;
  const int bidx = bx * NBY + by;

  // --- stage 0: halo tile load (32x32 = 4 loads/thread), OOB -> 0 ---
  const int gi0 = bx * SX - 1, gj0 = by * SY - 1;
  #pragma unroll
  for (int e = tid; e < 32 * 32; e += 256) {
    int tr = e >> 5, tc = e & 31;
    int gi = gi0 + tr, gj = gj0 + tc;
    float v = 0.0f;
    if (gi >= 0 && gi < H && gj >= 0 && gj < W) v = chm[gi * W + gj];
    tile[tr][tc] = v;
  }
  __syncthreads();

  // block-uniform parameters (uniform index -> scalar loads)
  const float th_b   = TH[bidx];
  const float favd_b = FAVD[bidx];
  const float mu     = fmaxf(__cosf(sza[bidx] * DEG), 1e-3f);
  const float kview  = -Gc * favd_b;    // exponent coeff, view direction
  const float ksun   = kview / mu;      // exponent coeff, sun direction

  // --- stage 1: per-pixel math, f32 partial sums for 7 channels + local max ---
  float s0=0.f,s1=0.f,s2=0.f,s3=0.f,s4=0.f,s5=0.f,s6=0.f;
  float mx = 0.0f;
  for (int p = tid; p < NPIX; p += 256) {
    int r = p / SY, c = p - r * SY;
    float v = tile[r+1][c+1];
    float m = v > 0.0f ? 1.0f : 0.0f;
    float ns =
      (tile[r  ][c] > 0.f?1.f:0.f) + (tile[r  ][c+1] > 0.f?1.f:0.f) + (tile[r  ][c+2] > 0.f?1.f:0.f) +
      (tile[r+1][c] > 0.f?1.f:0.f)                                  + (tile[r+1][c+2] > 0.f?1.f:0.f) +
      (tile[r+2][c] > 0.f?1.f:0.f) + (tile[r+2][c+1] > 0.f?1.f:0.f) + (tile[r+2][c+2] > 0.f?1.f:0.f);
    int gi = bx * SX + r, gj = by * SY + c;
    bool interior = (gi >= 1) & (gi < H-1) & (gj >= 1) & (gj < W-1);
    // inner = 8 - ns > 1  <=>  ns < 6.5 (ns is integer-valued)
    float edge = (m > 0.0f && interior && ns < 6.5f) ? 1.0f : 0.0f;
    float gap  = (m == 0.0f) ? 1.0f : (edge > 0.0f ? 0.5f : 0.0f); // (2 - tchm)/2
    float crown = fmaxf(v - th_b, 0.0f);
    float gs = __expf(ksun  * crown);   // gap_sun
    float gv = __expf(kview * crown);   // gap_view
    s0 += gs;        s1 += gv;      s2 += edge; s3 += v;
    s4 += edge * gs; s5 += m * gv;  s6 += gap;
    mx = fmaxf(mx, v);
  }
  part[0][tid]=s0; part[1][tid]=s1; part[2][tid]=s2; part[3][tid]=s3;
  part[4][tid]=s4; part[5][tid]=s5; part[6][tid]=s6;
  mred[tid] = mx;
  __syncthreads();

  // --- stage 2a: block max tree (uniform barriers, before divergent WMMA) ---
  #pragma unroll
  for (int s = 128; s > 0; s >>= 1) {
    if (tid < s) mred[tid] = fmaxf(mred[tid], mred[tid + s]);
    __syncthreads();
  }
  if (tid == 0) atomicMax((int*)ws, __float_as_int(mred[0]));

  // --- stage 2b: compact 256 partials -> 32 groups (8-way f32 pre-sum), f16 store ---
  #pragma unroll
  for (int e = tid; e < 16 * 32; e += 256) {
    int cch = e >> 5, k = e & 31;
    float sum = 0.0f;
    if (cch < 7) {
      #pragma unroll
      for (int j = 0; j < 8; ++j) sum += part[cch][k + 32 * j];
    }
    part2[cch][k] = (_Float16)sum;
  }
  __syncthreads();

  // --- stage 3: wave 0 reduces 16x32 via WMMA: D[m,n] = sum_k A[m,k] * 1 ---
  if (tid < 32) {
    const int lane = tid;
    const int row  = lane & 15;
    const int hi   = lane >> 4;
    // A-matrix 16-bit 16x32 layout: lanes 0-15 hold K=0..7,16..23; lanes 16-31 hold K=8..15,24..31
    const int base1 = hi ? 8  : 0;
    const int base2 = hi ? 24 : 16;
    v8h plo = *(v8h*)(&part2[row][base1]);
    v8h phi = *(v8h*)(&part2[row][base2]);
    v16h a;
    #pragma unroll
    for (int i = 0; i < 8; ++i) { a[i] = plo[i]; a[i + 8] = phi[i]; }
    v16h b;
    #pragma unroll
    for (int i = 0; i < 16; ++i) b[i] = (_Float16)1.0f;  // ones: layout-independent
    v8f cacc = {};
    v8f d = __builtin_amdgcn_wmma_f32_16x16x32_f16(
        /*neg_a=*/false, a, /*neg_b=*/false, b,
        /*c_mod=*/(short)0, cacc, /*reuse_a=*/false, /*reuse_b=*/false);

    // Lane 0, VGPR r holds D[M=r][N=0] = channel-r block sum
    if (tid == 0) {
      const float Ninv = 1.0f / (float)NPIX;
      float Pgs   = d[0] * Ninv;
      float Pgv   = d[1] * Ninv;
      float te7   = d[2] * Ninv;
      float mCHM  = d[3] * Ninv;       // mean(CHM); /max deferred to finalize
      float te11  = d[4] * Ninv;
      float te12  = d[5] * Ninv;
      float f_gap = d[6] * Ninv;
      float Pboth = Pgs * Pgv;
      float Kg = f_gap * Pgs;
      float Kz = f_gap * (1.0f - Pgs);
      float Kc = (1.0f - f_gap) * Pboth;
      float Kt = fmaxf((1.0f - f_gap) - Kc, 0.0f);
      float RL = rl[bidx], TL = tl[bidx], RS = rs[bidx], BE = belta[bidx];
      float hot = 1.0f + 0.1f * __cosf(saa[bidx] * DEG);
      float A = (RL*Kc + TL*BE*Kt + RS*Kg + RS*BE*Kz
                 + TL*(1.0f - BE)*te11 + RS*te12*f_gap) * hot;
      float B = RL * te7 * mCHM * hot;   // becomes  B / max(CHM)  in finalize
      ws[1 + bidx]        = A;
      ws[1 + NBLK + bidx] = B;
    }
  }
}

// Tiny epilogue: out[band][i] = A[i] + B[i]/max, broadcast over 4 bands.
__global__ __launch_bounds__(256) void finalize_kernel(const float* __restrict__ ws,
                                                       float* __restrict__ out) {
  int i = blockIdx.x * 256 + threadIdx.x;
  if (i < NBLK) {
    float mxv = ws[0];                    // float bits written via int atomicMax
    float v = ws[1 + i] + ws[1 + NBLK + i] / mxv;
    out[i]            = v;
    out[NBLK + i]     = v;
    out[2 * NBLK + i] = v;
    out[3 * NBLK + i] = v;
  }
}

extern "C" void kernel_launch(void* const* d_in, const int* in_sizes, int n_in,
                              void* d_out, int out_size, void* d_ws, size_t ws_size,
                              hipStream_t stream) {
  // setup_inputs order: CHM, PATH1, PATH2, TH, FAVD, sza, saa, rl, tl, rs, belta
  const float* CHM   = (const float*)d_in[0];
  // PATH1 (d_in[1]) and PATH2 (d_in[2]) only feed te[2],te[3],te[6] -> unused in brf
  const float* TH    = (const float*)d_in[3];
  const float* FAVD  = (const float*)d_in[4];
  const float* sza   = (const float*)d_in[5];
  const float* saa   = (const float*)d_in[6];
  const float* rl    = (const float*)d_in[7];
  const float* tl    = (const float*)d_in[8];
  const float* rs    = (const float*)d_in[9];
  const float* belta = (const float*)d_in[10];
  float* out = (float*)d_out;
  float* ws  = (float*)d_ws;

  ws_init_kernel<<<1, 256, 0, stream>>>(ws);
  brf_kernel<<<dim3(NBX, NBY), 256, 0, stream>>>(CHM, TH, FAVD, sza, saa,
                                                 rl, tl, rs, belta, ws);
  finalize_kernel<<<(NBLK + 255) / 256, 256, 0, stream>>>(ws, out);
}
